// Attention_22522808500477
// MI455X (gfx1250) — matrix operand order repaired and checked by reading, not compiled, not run on hardware
//
#include <hip/hip_runtime.h>
#include <math.h>

typedef __attribute__((ext_vector_type(16))) __bf16 bf16x16;
typedef __attribute__((ext_vector_type(8)))  float  v8f;

#define B_   256
#define P_   196
#define ENC_ 2048
#define DEC_ 512
#define ATT_ 512
#define M_TOT (B_ * P_)          // 50176

// float -> bf16 bits, round-to-nearest-even
__device__ __forceinline__ unsigned short f2bf_bits(float f) {
    unsigned int u = __float_as_uint(f);
    u += 0x7FFFu + ((u >> 16) & 1u);
    return (unsigned short)(u >> 16);
}
__device__ __forceinline__ __bf16 f2bf(float f) {
    unsigned short s = f2bf_bits(f);
    return __builtin_bit_cast(__bf16, s);
}

// ---------------------------------------------------------------------------
// K0: W_enc [2048,512] f32 (row-major [K,N]) -> W_encT [512,2048] bf16 ([N,K])
// ---------------------------------------------------------------------------
__global__ __launch_bounds__(256) void wenc_transpose_kernel(
    const float* __restrict__ w, unsigned short* __restrict__ wT) {
    int idx = blockIdx.x * 256 + threadIdx.x;   // 0 .. 512*2048-1 over output
    int n = idx >> 11;                          // / 2048
    int k = idx & 2047;
    wT[idx] = f2bf_bits(w[(size_t)k * ATT_ + n]);
}

// ---------------------------------------------------------------------------
// K1: att2p[b][n] = dh[b,:] @ W_dec[:,n] + b_dec[n] + b_enc[n]
// 32 blocks x 256 threads; each block does 8 batch rows, thread covers n and n+256
// ---------------------------------------------------------------------------
__global__ __launch_bounds__(256) void att2_kernel(
    const float* __restrict__ dh, const float* __restrict__ wdec,
    const float* __restrict__ benc, const float* __restrict__ bdec,
    float* __restrict__ att2p) {
    const int t  = threadIdx.x;
    const int b0 = blockIdx.x * 8;
    float acc[8][2];
#pragma unroll
    for (int j = 0; j < 8; ++j) { acc[j][0] = 0.f; acc[j][1] = 0.f; }
    for (int k = 0; k < DEC_; ++k) {
        float w0 = wdec[(size_t)k * ATT_ + t];
        float w1 = wdec[(size_t)k * ATT_ + t + 256];
#pragma unroll
        for (int j = 0; j < 8; ++j) {
            float h = dh[(size_t)(b0 + j) * DEC_ + k];
            acc[j][0] = fmaf(h, w0, acc[j][0]);
            acc[j][1] = fmaf(h, w1, acc[j][1]);
        }
    }
    float bias0 = benc[t] + bdec[t];
    float bias1 = benc[t + 256] + bdec[t + 256];
#pragma unroll
    for (int j = 0; j < 8; ++j) {
        att2p[(size_t)(b0 + j) * ATT_ + t]       = acc[j][0] + bias0;
        att2p[(size_t)(b0 + j) * ATT_ + t + 256] = acc[j][1] + bias1;
    }
}

// ---------------------------------------------------------------------------
// K2: fused score kernel.
// Block = 256 threads = 8 waves; computes 32 rows of att1 across all N=512,
// then epilogue: relu(acc + att2p) . W_full + b_full -> att[m]
// Wave w: both 16-row M-tiles, N-tiles {4w .. 4w+3} (64 columns).
// ---------------------------------------------------------------------------
__global__ __launch_bounds__(256) void att_score_kernel(
    const float* __restrict__ enc, const unsigned short* __restrict__ wTbits,
    const float* __restrict__ att2p, const float* __restrict__ wfull,
    const float* __restrict__ bfull, float* __restrict__ att) {
    __shared__ float smem[32];
    const __bf16* __restrict__ wT = reinterpret_cast<const __bf16*>(wTbits);

    const int tid    = threadIdx.x;
    const int wave   = tid >> 5;
    const int lane   = tid & 31;
    const int lane16 = lane & 15;
    const int half   = (lane >> 4) & 1;
    const long long block_m = (long long)blockIdx.x * 32;
    const int nbase = wave * 64;

    if (tid < 32) smem[tid] = 0.0f;
    __syncthreads();

    v8f acc[2][4];
#pragma unroll
    for (int mt = 0; mt < 2; ++mt)
#pragma unroll
        for (int t = 0; t < 4; ++t)
            acc[mt][t] = (v8f){0.f, 0.f, 0.f, 0.f, 0.f, 0.f, 0.f, 0.f};

    for (int ks = 0; ks < ENC_ / 32; ++ks) {
        const int k0 = ks * 32;
        // ---- A fragments: 16x32 fp32 tile -> bf16 regs, contiguous k order (same as B):
        // lanes 0-15 hold row M=lane, K=0..15; lanes 16-31 K=16..31
        bf16x16 afrag[2];
#pragma unroll
        for (int mt = 0; mt < 2; ++mt) {
            const float* ap = enc + (size_t)(block_m + mt * 16 + lane16) * ENC_;
            const int ka = k0 + half * 16;
            float4 c0 = *(const float4*)(ap + ka);
            float4 c1 = *(const float4*)(ap + ka + 4);
            float4 c2 = *(const float4*)(ap + ka + 8);
            float4 c3 = *(const float4*)(ap + ka + 12);
            bf16x16 f;
            f[0]  = f2bf(c0.x); f[1]  = f2bf(c0.y); f[2]  = f2bf(c0.z); f[3]  = f2bf(c0.w);
            f[4]  = f2bf(c1.x); f[5]  = f2bf(c1.y); f[6]  = f2bf(c1.z); f[7]  = f2bf(c1.w);
            f[8]  = f2bf(c2.x); f[9]  = f2bf(c2.y); f[10] = f2bf(c2.z); f[11] = f2bf(c2.w);
            f[12] = f2bf(c3.x); f[13] = f2bf(c3.y); f[14] = f2bf(c3.z); f[15] = f2bf(c3.w);
            afrag[mt] = f;
        }
        // ---- B fragments from pre-transposed bf16 W_encT[N][K]:
        // lane = column N; lanes 0-15 get K=k0..k0+15, lanes 16-31 K=k0+16..k0+31
        bf16x16 bfrag[4];
#pragma unroll
        for (int t = 0; t < 4; ++t) {
            const int n = nbase + t * 16 + lane16;
            bfrag[t] = *(const bf16x16*)(wT + (size_t)n * ENC_ + k0 + half * 16);
        }
        // ---- 8 WMMAs per k-step
#pragma unroll
        for (int mt = 0; mt < 2; ++mt)
#pragma unroll
            for (int t = 0; t < 4; ++t)
                acc[mt][t] = __builtin_amdgcn_wmma_f32_16x16x32_bf16(
                    false, afrag[mt], false, bfrag[t],
                    (short)0, acc[mt][t], false, false);
    }

    // ---- Epilogue: relu(acc + att2p[b,n]) * W_full[n], reduce over n ----
    // C/D layout: VGPR r, lanes 0-15 -> (M=r, N=lane); lanes 16-31 -> (M=8+r)
#pragma unroll
    for (int mt = 0; mt < 2; ++mt) {
#pragma unroll
        for (int r = 0; r < 8; ++r) {
            const int m_local = mt * 16 + r + 8 * half;
            const long long m_g = block_m + m_local;
            const int b = (int)(m_g / P_);
            float partial = 0.f;
#pragma unroll
            for (int t = 0; t < 4; ++t) {
                const int n = nbase + t * 16 + lane16;
                float v = acc[mt][t][r] + att2p[(size_t)b * ATT_ + n];
                v = fmaxf(v, 0.f);
                partial = fmaf(v, wfull[n], partial);
            }
            // reduce across the 16-lane half (each half owns one m)
            partial += __shfl_xor(partial, 1, 32);
            partial += __shfl_xor(partial, 2, 32);
            partial += __shfl_xor(partial, 4, 32);
            partial += __shfl_xor(partial, 8, 32);
            if (lane16 == 0) atomicAdd(&smem[m_local], partial);
        }
    }
    __syncthreads();
    if (tid < 32) att[block_m + tid] = smem[tid] + bfull[0];
}

// ---------------------------------------------------------------------------
// K3: softmax over P=196 per batch row
// ---------------------------------------------------------------------------
__global__ __launch_bounds__(256) void softmax_kernel(
    const float* __restrict__ att, float* __restrict__ alpha) {
    __shared__ float red[256];
    const int b = blockIdx.x, t = threadIdx.x;
    float v = (t < P_) ? att[(size_t)b * P_ + t] : -3.402823466e38f;
    red[t] = v;
    __syncthreads();
    for (int s = 128; s > 0; s >>= 1) {
        if (t < s) red[t] = fmaxf(red[t], red[t + s]);
        __syncthreads();
    }
    float mx = red[0];
    __syncthreads();
    float e = (t < P_) ? __expf(v - mx) : 0.f;
    red[t] = e;
    __syncthreads();
    for (int s = 128; s > 0; s >>= 1) {
        if (t < s) red[t] += red[t + s];
        __syncthreads();
    }
    float inv = 1.0f / red[0];
    if (t < P_) alpha[(size_t)b * P_ + t] = e * inv;
}

// ---------------------------------------------------------------------------
// K4: context[b,e] = sum_p alpha[b,p] * enc[b,p,e]  (pure bandwidth)
// Block per b; thread owns 8 consecutive e's (two float4's).
// ---------------------------------------------------------------------------
__global__ __launch_bounds__(256) void context_kernel(
    const float* __restrict__ enc, const float* __restrict__ alpha,
    float* __restrict__ out) {
    const int b = blockIdx.x, t = threadIdx.x;
    const float* ep = enc + (size_t)b * P_ * ENC_ + t * 8;
    float4 s0 = {0.f, 0.f, 0.f, 0.f}, s1 = {0.f, 0.f, 0.f, 0.f};
    for (int p = 0; p < P_; ++p) {
        float a = alpha[(size_t)b * P_ + p];
        const float4* q = (const float4*)(ep + (size_t)p * ENC_);
        if (p + 4 < P_) __builtin_prefetch(ep + (size_t)(p + 4) * ENC_, 0, 0);
        float4 x = q[0], y = q[1];
        s0.x = fmaf(a, x.x, s0.x); s0.y = fmaf(a, x.y, s0.y);
        s0.z = fmaf(a, x.z, s0.z); s0.w = fmaf(a, x.w, s0.w);
        s1.x = fmaf(a, y.x, s1.x); s1.y = fmaf(a, y.y, s1.y);
        s1.z = fmaf(a, y.z, s1.z); s1.w = fmaf(a, y.w, s1.w);
    }
    float* op = out + (size_t)b * ENC_ + t * 8;
    *(float4*)op = s0;
    *((float4*)op + 1) = s1;
}

// ---------------------------------------------------------------------------
extern "C" void kernel_launch(void* const* d_in, const int* in_sizes, int n_in,
                              void* d_out, int out_size, void* d_ws, size_t ws_size,
                              hipStream_t stream) {
    const float* enc   = (const float*)d_in[0];  // [256,196,2048]
    const float* dh    = (const float*)d_in[1];  // [256,512]
    const float* wenc  = (const float*)d_in[2];  // [2048,512]
    const float* benc  = (const float*)d_in[3];  // [512]
    const float* wdec  = (const float*)d_in[4];  // [512,512]
    const float* bdec  = (const float*)d_in[5];  // [512]
    const float* wfull = (const float*)d_in[6];  // [512]
    const float* bfull = (const float*)d_in[7];  // [1]

    float* out   = (float*)d_out;
    float* ctx   = out;                          // [256,2048]
    float* alpha = out + (size_t)B_ * ENC_;      // [256,196]

    char* ws = (char*)d_ws;
    unsigned short* wT  = (unsigned short*)ws;                                // 2 MB
    float* att2p        = (float*)(ws + (size_t)2 * 1024 * 1024);             // 512 KB
    float* att          = (float*)(ws + (size_t)2 * 1024 * 1024 + 512 * 1024);// 200 KB

    wenc_transpose_kernel<<<(ATT_ * ENC_) / 256, 256, 0, stream>>>(wenc, wT);
    att2_kernel<<<B_ / 8, 256, 0, stream>>>(dh, wdec, benc, bdec, att2p);
    att_score_kernel<<<M_TOT / 32, 256, 0, stream>>>(enc, wT, att2p, wfull, bfull, att);
    softmax_kernel<<<B_, 256, 0, stream>>>(att, alpha);
    context_kernel<<<B_, 256, 0, stream>>>(enc, alpha, ctx);
}
